// QuantizedLinear_88270167867494
// MI455X (gfx1250) — compile-verified, matrix operand order checked by reading, not run on hardware
//
#include <hip/hip_runtime.h>

#define OUT_F 11008
#define IN_F  4096
#define M_TOTAL 8192

#define BM 256
#define BN 128
#define KC 64
#define AST 72   // LDS row stride in halves (64 + 8 pad -> conflict-free ds_load_b128)
#define BST 72
#define NSTAGES (IN_F / KC)   // 64 K-chunks

#define GLOBAL_AS __attribute__((address_space(1)))
#define LDS_AS    __attribute__((address_space(3)))

typedef _Float16 v4h  __attribute__((ext_vector_type(4)));
typedef _Float16 v8h  __attribute__((ext_vector_type(8)));
typedef _Float16 v16h __attribute__((ext_vector_type(16)));
typedef float    v8f  __attribute__((ext_vector_type(8)));
typedef int      v4i  __attribute__((ext_vector_type(4)));

// ---------- pass 1: x fp32 -> fp16 (one-shot HBM read, L2-resident result) ----------
__global__ __launch_bounds__(256) void cvt_x_kernel(const float* __restrict__ x,
                                                    _Float16* __restrict__ xh) {
  int i = blockIdx.x * 256 + threadIdx.x;
  float4 f = ((const float4*)x)[i];
  v4h h;
  h[0] = (_Float16)f.x; h[1] = (_Float16)f.y;
  h[2] = (_Float16)f.z; h[3] = (_Float16)f.w;
  ((v4h*)xh)[i] = h;
}

// ---------- pass 2: int32-per-byte -> dense packed nibbles (4x smaller) ----------
__global__ __launch_bounds__(256) void pack_w_kernel(const int* __restrict__ q,
                                                     unsigned int* __restrict__ wp) {
  int i = blockIdx.x * 256 + threadIdx.x;
  int4 v = ((const int4*)q)[i];
  unsigned int b = (unsigned int)(v.x & 255)
                 | ((unsigned int)(v.y & 255) << 8)
                 | ((unsigned int)(v.z & 255) << 16)
                 | ((unsigned int)(v.w & 255) << 24);
  wp[i] = b;
}

// 16B global -> LDS copy; async (ASYNCcnt-tracked, no VGPR round trip) when available.
__device__ __forceinline__ void a_chunk_to_lds(const _Float16* __restrict__ src,
                                               _Float16* dst) {
#if __has_builtin(__builtin_amdgcn_global_load_async_to_lds_b128)
  __builtin_amdgcn_global_load_async_to_lds_b128((GLOBAL_AS v4i*)src,
                                                 (LDS_AS v4i*)dst, 0, 0);
#else
  *(uint4*)dst = *(const uint4*)src;
#endif
}

// ---------- pass 3: WMMA f16 GEMM, double-buffered LDS, async A-tile copies ----------
__global__ __launch_bounds__(256) void qlinear_gemm_kernel(
    const _Float16* __restrict__ xh,        // [M_TOTAL, IN_F] row-major f16
    const unsigned char* __restrict__ wp,   // [OUT_F, IN_F/2] packed nibbles
    const float* __restrict__ scale,        // scalar
    const float* __restrict__ bias,         // [OUT_F]
    float* __restrict__ out)                // [M_TOTAL, OUT_F]
{
  __shared__ _Float16 sA[2 * BM * AST];   // 73,728 B
  __shared__ _Float16 sB[2 * BN * BST];   // 36,864 B

  const int tid  = threadIdx.x;
  const int lane = tid & 31;
  const int wave = tid >> 5;
  const int wm   = (wave & 3) * 64;   // 4 waves along M
  const int wn   = (wave >> 2) * 64;  // 2 waves along N
  const int n0   = blockIdx.x * BN;
  const int m0   = blockIdx.y * BM;

  const int r16 = lane & 15;          // fragment row/col within 16
  const int kh  = (lane >> 4) << 3;   // K-half select per ISA 16-bit layout

  // Issue the 32 KB A tile for K-chunk at `kc` into buffer `buf` (8 x b128 per thread).
  auto issueA = [&](int kc, int buf) {
#pragma unroll
    for (int i = 0; i < 8; ++i) {
      int c   = i * 256 + tid;        // 2048 chunks of 16B, coalesced 128B per row
      int row = c >> 3;
      int kch = c & 7;
      a_chunk_to_lds(xh + (size_t)(m0 + row) * IN_F + kc + kch * 8,
                     &sA[buf * (BM * AST) + row * AST + kch * 8]);
    }
  };
  auto loadB = [&](int kc) -> uint4 {   // 16B packed nibbles = 32 K-values per thread
    int row = tid >> 1, half = tid & 1;
    return *(const uint4*)(wp + (size_t)(n0 + row) * (IN_F / 2) + (kc >> 1) + half * 16);
  };
  auto storeB = [&](uint4 u, int buf) { // dequant int4 -> f16 in registers, 4 x b128 to LDS
    int row = tid >> 1, half = tid & 1;
    unsigned int uw[4] = {u.x, u.y, u.z, u.w};
#pragma unroll
    for (int w = 0; w < 4; ++w) {
      v8h o;
#pragma unroll
      for (int b = 0; b < 4; ++b) {
        int byte = (int)((uw[w] >> (8 * b)) & 0xFF);
        o[2 * b]     = (_Float16)((byte & 15) - 8);  // low nibble = even K
        o[2 * b + 1] = (_Float16)((byte >> 4) - 8);  // high nibble = odd K
      }
      *(v8h*)(&sB[buf * (BN * BST) + row * BST + half * 32 + w * 8]) = o;
    }
  };

  v8f acc[4][4];
#pragma unroll
  for (int i = 0; i < 4; ++i)
#pragma unroll
    for (int j = 0; j < 4; ++j) {
      v8f z = {};
      acc[i][j] = z;
    }

  // ---- prologue: stage 0 in flight ----
  issueA(0, 0);
  storeB(loadB(0), 0);

  for (int t = 0; t < NSTAGES; ++t) {
    const int cur = t & 1;
    const int nxt = cur ^ 1;
    const bool notLast = (t + 1 < NSTAGES);

    // B packed load targets VGPRs only (no LDS hazard) -> issue it BEFORE the
    // barrier so it has barrier latency + the whole compute phase to complete.
    uint4 bNext{};
    if (notLast) bNext = loadB((t + 1) * KC);

    // Our stage-`cur` async A copies must have landed before the barrier.
    asm volatile("s_wait_asynccnt 0" ::: "memory");
    __syncthreads();  // everyone's A-cur (async) and B-cur (ds_store, DScnt-flushed) visible

    if (notLast) {
      issueA((t + 1) * KC, nxt);        // async LDS writes: legal only after the barrier
    }

    const _Float16* aBase = &sA[cur * (BM * AST)];
    const _Float16* bBase = &sB[cur * (BN * BST)];
#pragma unroll
    for (int kk = 0; kk < KC; kk += 32) {
      v16h bf[4];
#pragma unroll
      for (int j = 0; j < 4; ++j) {
        const _Float16* p = bBase + (wn + 16 * j + r16) * BST + kk + kh;
        v8h lo = *(const v8h*)p;
        v8h hi = *(const v8h*)(p + 16);
        bf[j] = __builtin_shufflevector(lo, hi, 0,1,2,3,4,5,6,7,8,9,10,11,12,13,14,15);
      }
#pragma unroll
      for (int i = 0; i < 4; ++i) {
        const _Float16* p = aBase + (wm + 16 * i + r16) * AST + kk + kh;
        v8h lo = *(const v8h*)p;
        v8h hi = *(const v8h*)(p + 16);
        v16h af = __builtin_shufflevector(lo, hi, 0,1,2,3,4,5,6,7,8,9,10,11,12,13,14,15);
#pragma unroll
        for (int j = 0; j < 4; ++j) {
          acc[i][j] = __builtin_amdgcn_wmma_f32_16x16x32_f16(
              false, af, false, bf[j], (short)0, acc[i][j], false, false);
        }
      }
    }

    if (notLast) storeB(bNext, nxt);    // safe: all readers of `nxt` passed this iter's barrier
  }

  // ---- epilogue: out = acc*scale + bias, per ISA f32 C/D layout ----
  const float s = scale[0];
#pragma unroll
  for (int j = 0; j < 4; ++j) {
    int col  = n0 + wn + 16 * j + r16;
    float bj = bias[col];
#pragma unroll
    for (int i = 0; i < 4; ++i) {
      int row0 = m0 + wm + 16 * i + ((lane >> 4) << 3);
      float* op = out + (size_t)row0 * OUT_F + col;
#pragma unroll
      for (int v = 0; v < 8; ++v) {
        op[(size_t)v * OUT_F] = acc[i][j][v] * s + bj;
      }
    }
  }
}

extern "C" void kernel_launch(void* const* d_in, const int* in_sizes, int n_in,
                              void* d_out, int out_size, void* d_ws, size_t ws_size,
                              hipStream_t stream) {
  const float* x     = (const float*)d_in[0];
  const int*   q     = (const int*)d_in[1];
  const float* scale = (const float*)d_in[2];
  const float* bias  = (const float*)d_in[3];
  float*       out   = (float*)d_out;

  // workspace layout: [ x_f16 : 67,108,864 B ][ packed W : 22,544,384 B ]
  _Float16*      xh = (_Float16*)d_ws;
  unsigned char* wp = (unsigned char*)d_ws + (size_t)M_TOTAL * IN_F * 2;

  cvt_x_kernel<<<(M_TOTAL * IN_F / 4) / 256, 256, 0, stream>>>(x, xh);
  pack_w_kernel<<<(OUT_F * IN_F / 2 / 4) / 256, 256, 0, stream>>>(q, (unsigned int*)wp);

  dim3 grid(OUT_F / BN, M_TOTAL / BM);  // (86, 32)
  qlinear_gemm_kernel<<<grid, 256, 0, stream>>>(xh, wp, scale, bias, out);
}
